// SubstitutionEmbedding_18786186953089
// MI455X (gfx1250) — matrix operand order repaired and checked by reading, not compiled, not run on hardware
//
#include <hip/hip_runtime.h>

typedef _Float16 half_t;
typedef __attribute__((ext_vector_type(16))) _Float16 v16h;
typedef __attribute__((ext_vector_type(8)))  _Float16 v8h;
typedef __attribute__((ext_vector_type(4)))  _Float16 v4h;
typedef __attribute__((ext_vector_type(8)))  float    v8f;
typedef int v4i_t __attribute__((vector_size(16)));

#define B_     32
#define L1_    8192
#define L2_    32768
#define LTOT_  40960
#define C_     32      // CONV_DEPTH
#define S_     8
#define E_     256     // EMBED_DIM
#define K_     256     // GEMM K = S_*C_
#define M2_    (B_*4096)   // conv2 output rows
#define M1_    (B_*1024)   // conv1 output rows

// ---------------------------------------------------------------------------
// gfx1250 async global->LDS copy (ASYNCcnt-tracked, bypasses VGPRs).
// Builtin signature (from compiler diagnostic): pointee type is int4-vector.
// LDS operand: low 32 bits of the generic shared pointer == LDS offset.
// ---------------------------------------------------------------------------
#if defined(__has_builtin)
#if __has_builtin(__builtin_amdgcn_global_load_async_to_lds_b128)
#define HAVE_ASYNC_B128 1
#endif
#if __has_builtin(__builtin_amdgcn_s_wait_asynccnt)
#define HAVE_WAIT_ASYNC 1
#endif
#endif

__device__ __forceinline__ void async_copy_b128(const void* g, void* l) {
#ifdef HAVE_ASYNC_B128
  __builtin_amdgcn_global_load_async_to_lds_b128(
      (__attribute__((address_space(1))) v4i_t*)(uintptr_t)g,
      (__attribute__((address_space(3))) v4i_t*)(uint32_t)(uintptr_t)l,
      0, 0);
#else
  asm volatile("global_load_async_to_lds_b128 %0, %1, off"
               :: "v"((uint32_t)(uintptr_t)l), "v"(g) : "memory");
#endif
}

__device__ __forceinline__ void async_wait0() {
#ifdef HAVE_WAIT_ASYNC
  __builtin_amdgcn_s_wait_asynccnt(0);
#else
  asm volatile("s_wait_asynccnt 0x0" ::: "memory");
#endif
}

// ---------------------------------------------------------------------------
// Embedding: e[c] = ve[val][c] + de[dep][c] + se[0][p0][c] + se[1][p1][c] + se[2][p2][c]
// 32 channels, computed in f32, stored as f16 (to LDS or global).
// se layout: (3, 128, 32).
// ---------------------------------------------------------------------------
__device__ __forceinline__ void embed_store(const float* __restrict__ ve,
                                            const float* __restrict__ de,
                                            const float* __restrict__ se,
                                            int val, int dep, int p0, int p1, int p2,
                                            half_t* dst) {
  const float4* a = (const float4*)(ve + val * C_);
  const float4* b = (const float4*)(de + dep * C_);
  const float4* c = (const float4*)(se + p0 * C_);
  const float4* d = (const float4*)(se + (128 + p1) * C_);
  const float4* e = (const float4*)(se + (256 + p2) * C_);
#pragma unroll
  for (int q = 0; q < 8; ++q) {
    float4 A = a[q], Bv = b[q], Cv = c[q], Dv = d[q], Ev = e[q];
    v4h h;
    h[0] = (half_t)(A.x + Bv.x + Cv.x + Dv.x + Ev.x);
    h[1] = (half_t)(A.y + Bv.y + Cv.y + Dv.y + Ev.y);
    h[2] = (half_t)(A.z + Bv.z + Cv.z + Dv.z + Ev.z);
    h[3] = (half_t)(A.w + Bv.w + Cv.w + Dv.w + Ev.w);
    *(v4h*)(dst + q * 4) = h;
  }
}

// ---------------------------------------------------------------------------
// WMMA fragment loaders (wave32, 16x16x32 f16).
// A (16x32, MxK): lane<16 -> row=lane, kb=0; lane>=16 -> row=lane-16, kb=16;
//                 halves = {kb..kb+7, kb+8..kb+15}. Two contiguous 16B reads.
// B (32x16, KxN): col = lane&15, K = (lane>>4)*16 + h. One contiguous 32B read
//                 when W is staged [N][K] in LDS.
// C/D (16x16 f32): VGPR r -> row = r + (lane>>4)*8, col = lane&15.
// ---------------------------------------------------------------------------
__device__ __forceinline__ v16h load_frag_a(const half_t* At, int row, int ks, int lane) {
  int kb = ks * 32 + ((lane >> 4) << 4);
  v8h lo = *(const v8h*)(At + row * K_ + kb);
  v8h hi = *(const v8h*)(At + row * K_ + kb + 8);
  v16h r;
#pragma unroll
  for (int i = 0; i < 8; ++i) { r[i] = lo[i]; r[i + 8] = hi[i]; }
  return r;
}

__device__ __forceinline__ v16h load_frag_b(const half_t* Wt, int ncol, int ks, int lane) {
  int kb = ks * 32 + ((lane >> 4) << 4);
  return *(const v16h*)(Wt + ncol * K_ + kb);
}

// ---------------------------------------------------------------------------
// Kernel 1: fused part-2 embedding + conv2 GEMM.  M=131072, N=32, K=256.
// Block: 256 threads = 8 waves, tile 64(M) x 32(N); wave w -> mtile=w>>1, n0=(w&1)*16.
// Writes y f16: y[(b*4096+j)*32 + o].
// ---------------------------------------------------------------------------
__global__ void conv2_embed_wmma(const int* __restrict__ value, const int* __restrict__ depth,
                                 const int* __restrict__ pos,
                                 const float* __restrict__ ve2, const float* __restrict__ de2,
                                 const float* __restrict__ se2,
                                 const float* __restrict__ w2, const float* __restrict__ b2,
                                 half_t* __restrict__ y) {
  __shared__ __align__(128) half_t At[64 * K_];   // 32 KB
  __shared__ __align__(128) half_t Wt[C_ * K_];   // 16 KB, [o][k]
  const int tid = threadIdx.x;
  const int m0  = blockIdx.x * 64;          // 2048 blocks; 4096%64==0 -> one batch/block
  const int b   = m0 >> 12;
  const int j0  = m0 & 4095;

  // Prefetch this block's index streams while we transpose W into LDS.
  {
    int gi0 = b * LTOT_ + L1_ + j0 * S_ + tid * 2;
    __builtin_prefetch(&pos[gi0 * 3], 0, 1);
    __builtin_prefetch(&value[gi0], 0, 1);
  }
  // Stage W2^T: Wt[o][k] = conv2_w[o][c][s], k = s*32 + c
  for (int e = tid; e < C_ * K_; e += 256) {
    int o = e >> 8, k = e & 255;
    Wt[e] = (half_t)w2[o * K_ + (k & 31) * S_ + (k >> 5)];
  }
  // Stage A: 512 consecutive part-2 tokens -> rows 0..63 (row = tt>>3, kcol = (tt&7)*32)
  const int tokbase = j0 * S_;
#pragma unroll
  for (int q = 0; q < 2; ++q) {
    int tt  = tid * 2 + q;
    int gi  = b * LTOT_ + L1_ + tokbase + tt;
    embed_store(ve2, de2, se2, value[gi], depth[gi],
                pos[gi * 3 + 0], pos[gi * 3 + 1], pos[gi * 3 + 2], At + tt * C_);
  }
  __syncthreads();

  const int w = tid >> 5, lane = tid & 31;
  const int mtile = w >> 1, n0 = (w & 1) * 16;
  const int row = mtile * 16 + (lane & 15);
  v8f acc = {};
#pragma unroll
  for (int ks = 0; ks < 8; ++ks) {
    v16h a  = load_frag_a(At, row, ks, lane);
    v16h bf = load_frag_b(Wt, n0 + (lane & 15), ks, lane);
    acc = __builtin_amdgcn_wmma_f32_16x16x32_f16(false, a, false, bf, (short)0, acc, false, false);
  }
  const int o    = n0 + (lane & 15);
  const float bv = b2[o];
  const int mloc = mtile * 16 + ((lane >> 4) << 3);
#pragma unroll
  for (int r = 0; r < 8; ++r) {
    int m = m0 + mloc + r;
    y[(size_t)m * C_ + o] = (half_t)(acc[r] + bv);
  }
}

// ---------------------------------------------------------------------------
// Kernel 2: per-batch compaction of y.  One 1024-thread block per batch row.
// sel[j] = val2[j*8] != 0; exclusive scan; scatter 64B rows; zero the tail.
// ---------------------------------------------------------------------------
__global__ void compact_scan(const int* __restrict__ value,
                             const half_t* __restrict__ y, half_t* __restrict__ yc) {
  __shared__ int ssum[1024];
  const int b = blockIdx.x, tid = threadIdx.x;
  const int* val2 = value + b * LTOT_ + L1_;
  int sel[4], incl[4], run = 0;
  const int jb = tid * 4;
#pragma unroll
  for (int q = 0; q < 4; ++q) {
    sel[q] = (val2[(jb + q) * S_] != 0) ? 1 : 0;
    run += sel[q]; incl[q] = run;
  }
  ssum[tid] = run;
  __syncthreads();
  for (int off = 1; off < 1024; off <<= 1) {
    int v = ssum[tid];
    int u = (tid >= off) ? ssum[tid - off] : 0;
    __syncthreads();
    ssum[tid] = v + u;
    __syncthreads();
  }
  const int excl  = ssum[tid] - run;
  const int total = ssum[1023];

  const uint4* src = (const uint4*)(y  + (size_t)b * 4096 * C_);
  uint4*       dst = (uint4*)      (yc + (size_t)b * 4096 * C_);
#pragma unroll
  for (int q = 0; q < 4; ++q) {
    if (sel[q]) {
      int j = jb + q, d = excl + incl[q] - 1;
#pragma unroll
      for (int u = 0; u < 4; ++u) dst[d * 4 + u] = src[j * 4 + u];
    }
  }
  uint4 z = {0u, 0u, 0u, 0u};
  for (int r = total + tid; r < 4096; r += 1024) {
#pragma unroll
    for (int u = 0; u < 4; ++u) dst[r * 4 + u] = z;
  }
}

// ---------------------------------------------------------------------------
// Kernel 3: per-batch scan of part-1 mask (val1==2) -> gather index (or -1).
// ---------------------------------------------------------------------------
__global__ void order_scan(const int* __restrict__ value, int* __restrict__ gidx) {
  __shared__ int ssum[1024];
  const int b = blockIdx.x, tid = threadIdx.x;
  const int* val1 = value + b * LTOT_;
  int mk[8], incl[8], run = 0;
  const int ib = tid * 8;
#pragma unroll
  for (int q = 0; q < 8; ++q) {
    mk[q] = (val1[ib + q] == 2) ? 1 : 0;
    run += mk[q]; incl[q] = run;
  }
  ssum[tid] = run;
  __syncthreads();
  for (int off = 1; off < 1024; off <<= 1) {
    int v = ssum[tid];
    int u = (tid >= off) ? ssum[tid - off] : 0;
    __syncthreads();
    ssum[tid] = v + u;
    __syncthreads();
  }
  const int excl = ssum[tid] - run;
#pragma unroll
  for (int q = 0; q < 8; ++q) {
    int o = excl + incl[q] - 1;
    if (o > 4095) o = 4095;
    gidx[b * L1_ + ib + q] = mk[q] ? o : -1;
  }
}

// ---------------------------------------------------------------------------
// Kernel 4: build X1 f16 (the conv1 A-matrix, contiguous X1[m*256+k]).
// Token = substituted y_c row (copy) or part-1 embedding.
// ---------------------------------------------------------------------------
__global__ void build_x(const int* __restrict__ value, const int* __restrict__ depth,
                        const int* __restrict__ pos,
                        const float* __restrict__ ve1, const float* __restrict__ de1,
                        const float* __restrict__ se1,
                        const int* __restrict__ gidx, const half_t* __restrict__ yc,
                        half_t* __restrict__ X1) {
  const int t = blockIdx.x * blockDim.x + threadIdx.x;   // token 0..B*L1-1
  const int b = t >> 13, i = t & (L1_ - 1);
  const int g = gidx[t];
  half_t* dst = X1 + (size_t)t * C_;
  if (g >= 0) {
    const uint4* src = (const uint4*)(yc + ((size_t)b * 4096 + g) * C_);
    uint4* d4 = (uint4*)dst;
#pragma unroll
    for (int u = 0; u < 4; ++u) d4[u] = src[u];
  } else {
    int gi = b * LTOT_ + i;
    embed_store(ve1, de1, se1, value[gi], depth[gi],
                pos[gi * 3 + 0], pos[gi * 3 + 1], pos[gi * 3 + 2], dst);
  }
}

// ---------------------------------------------------------------------------
// Kernel 5: conv1 GEMM.  M=32768, N=256, K=256.  Block tile 64x64, 8 waves,
// each wave: 16(M) x 32(N) -> 2 accumulators, 2 WMMAs per k-step.
// A tile (contiguous 32 KB) staged via async global->LDS (ASYNCcnt), which
// runs concurrently with the W transpose-gather into LDS.
// ---------------------------------------------------------------------------
__global__ void conv1_wmma(const half_t* __restrict__ X1, const float* __restrict__ w1,
                           const float* __restrict__ b1, float* __restrict__ out) {
  __shared__ __align__(128) half_t At[64 * K_];   // 32 KB
  __shared__ __align__(128) half_t Wt[64 * K_];   // 32 KB, [o][k]
  const int tid = threadIdx.x;
  const int m0 = blockIdx.x * 64;   // 512
  const int o0 = blockIdx.y * 64;   // 4

  // Kick off async A-tile copy: 64 rows * 512 B = 2048 x b128.
  {
    const char* srcb = (const char*)(X1 + (size_t)m0 * K_);
    char*       dstb = (char*)At;
    for (int u = tid; u < 2048; u += 256)
      async_copy_b128(srcb + (size_t)u * 16, dstb + (size_t)u * 16);
  }
  // Speculative prefetch of the next M-block's A tile into L2.
  if (m0 + 64 < M1_) __builtin_prefetch(X1 + (size_t)(m0 + 64) * K_ + tid * 64, 0, 1);

  // Meanwhile transpose-stage W1^T: Wt[o][k] = conv1_w[o0+o][c][s], k = s*32+c.
  for (int e = tid; e < 64 * K_; e += 256) {
    int o = e >> 8, k = e & 255;
    Wt[e] = (half_t)w1[(size_t)(o0 + o) * K_ + (k & 31) * S_ + (k >> 5)];
  }
  async_wait0();
  __syncthreads();

  const int w = tid >> 5, lane = tid & 31;
  const int mtile = w >> 1, n0 = (w & 1) * 32;
  const int row = mtile * 16 + (lane & 15);
  v8f acc0 = {}, acc1 = {};
#pragma unroll
  for (int ks = 0; ks < 8; ++ks) {
    v16h a   = load_frag_a(At, row, ks, lane);
    v16h bf0 = load_frag_b(Wt, n0 + (lane & 15), ks, lane);
    v16h bf1 = load_frag_b(Wt, n0 + 16 + (lane & 15), ks, lane);
    acc0 = __builtin_amdgcn_wmma_f32_16x16x32_f16(false, a, false, bf0, (short)0, acc0, false, false);
    acc1 = __builtin_amdgcn_wmma_f32_16x16x32_f16(false, a, false, bf1, (short)0, acc1, false, false);
  }
  const int col0 = o0 + n0 + (lane & 15);
  const int col1 = col0 + 16;
  const float bv0 = b1[col0], bv1 = b1[col1];
  const int mloc = mtile * 16 + ((lane >> 4) << 3);
#pragma unroll
  for (int r = 0; r < 8; ++r) {
    size_t m = (size_t)(m0 + mloc + r);
    out[m * E_ + col0] = acc0[r] + bv0;
    out[m * E_ + col1] = acc1[r] + bv1;
  }
}

// ---------------------------------------------------------------------------
extern "C" void kernel_launch(void* const* d_in, const int* in_sizes, int n_in,
                              void* d_out, int out_size, void* d_ws, size_t ws_size,
                              hipStream_t stream) {
  (void)in_sizes; (void)n_in; (void)out_size; (void)ws_size;
  const int*   value = (const int*)d_in[0];
  const int*   depth = (const int*)d_in[1];
  const int*   pos   = (const int*)d_in[2];
  const float* ve1   = (const float*)d_in[5];
  const float* de1   = (const float*)d_in[6];
  const float* se1   = (const float*)d_in[7];
  const float* ve2   = (const float*)d_in[8];
  const float* de2   = (const float*)d_in[9];
  const float* se2   = (const float*)d_in[10];
  const float* w1    = (const float*)d_in[11];
  const float* b1    = (const float*)d_in[12];
  const float* w2    = (const float*)d_in[13];
  const float* b2    = (const float*)d_in[14];
  float* out = (float*)d_out;

  char* ws = (char*)d_ws;
  half_t* y_ws = (half_t*)(ws);                          // 8 MB  (B*4096*32 f16)
  half_t* y_c  = (half_t*)(ws + (8u  << 20));            // 8 MB
  int*    gidx = (int*)   (ws + (16u << 20));            // 1 MB  (B*8192 i32)
  half_t* X1   = (half_t*)(ws + (17u << 20));            // 16 MB (B*8192*32 f16)

  conv2_embed_wmma<<<M2_ / 64, 256, 0, stream>>>(value, depth, pos, ve2, de2, se2, w2, b2, y_ws);
  compact_scan<<<B_, 1024, 0, stream>>>(value, y_ws, y_c);
  order_scan<<<B_, 1024, 0, stream>>>(value, gidx);
  build_x<<<(B_ * L1_) / 256, 256, 0, stream>>>(value, depth, pos, ve1, de1, se1, gidx, y_c, X1);
  conv1_wmma<<<dim3(M1_ / 64, E_ / 64, 1), 256, 0, stream>>>(X1, w1, b1, out);
}
